// JBF_53936199303803
// MI455X (gfx1250) — matrix operand order repaired and checked by reading, not compiled, not run on hardware
//
#include <hip/hip_runtime.h>

// ---------------------------------------------------------------------------
// Joint bilateral filter network for MI455X (gfx1250), wave32 + WMMA bf16.
// GEMM mapping: M = output channels (A = pre-swizzled weights, staged into
// LDS by the Tensor Data Mover), N = pixels (B = activations, raw NHWC bf16).
// D gives each lane 8 consecutive channels of one pixel -> single b128 stores.
// A fragments are double-buffered so each WMMA has the next fragment's
// ds_load_b128 pair in flight behind it (s_wait_dscnt <= 2 instead of 0).
// ---------------------------------------------------------------------------

#define BATCH 2
#define CHN   64
#define HH    256
#define WW    256
#define HW    (HH * WW)

typedef __attribute__((ext_vector_type(16))) __bf16        v16bf;
typedef __attribute__((ext_vector_type(8)))  float         v8f;
typedef __attribute__((ext_vector_type(4)))  unsigned int  v4u;
typedef __attribute__((ext_vector_type(4)))  unsigned int  u32x4;
typedef __attribute__((ext_vector_type(8)))  int           i32x8;
typedef __attribute__((ext_vector_type(4)))  int           i32x4;

union Frag16 { v16bf bf; v4u u[2]; };

#define WMMA_BF16(A, B, C) \
    __builtin_amdgcn_wmma_f32_16x16x32_bf16(false, (A).bf, false, (B).bf, (short)0, (C), false, false)

// branch-free round-to-nearest-even f32 -> bf16 bits (low 16 of result)
__device__ __forceinline__ unsigned int f2bf_u(float f) {
    union { float f; unsigned int u; } x; x.f = f;
    return (x.u + 0x7fffu + ((x.u >> 16) & 1u)) >> 16;
}

// ---------------------------------------------------------------------------
// Stage ndwords of weight fragments from global into LDS.
// Fast path: Tensor Data Mover (1-D tile descriptor per ISA cdna5 ch.8):
//   group0: [1:0]=count=1, [63:32]=lds_addr, [120:64]=global_addr, [127:126]=2
//   group1: data_size=3 (8B), tensor_dim0=tile_dim0=n8, tensor_dim1=tile_dim1=1,
//           tensor_dim0_stride=n8; groups 2/3 zero (<=2-D tensor).
// Only wave 0 issues the DMA (TDM ignores EXEC), waits TENSORcnt, then the
// workgroup barrier publishes the data.
// ---------------------------------------------------------------------------
__device__ __forceinline__ void stage_weights(unsigned int* lds,
                                              const unsigned int* gsrc,
                                              unsigned ndwords) {
#if defined(__HIP_DEVICE_COMPILE__) && __has_builtin(__builtin_amdgcn_tensor_load_to_lds)
    if (threadIdx.x < 32) {                       // wave-uniform branch
        unsigned n8 = ndwords >> 1;               // 8-byte elements
        unsigned long long ga = (unsigned long long)(size_t)gsrc;
        unsigned ldsoff = (unsigned)(size_t)lds;  // LDS aperture: low 32 bits = offset
        u32x4 g0 = { 1u,                                          // count=1
                     ldsoff,                                      // lds_addr
                     (unsigned)ga,                                // global_addr[31:0]
                     (unsigned)((ga >> 32) & 0x01ffffffu) | 0x80000000u }; // addr[56:32] | type=2
        i32x8 g1 = { (int)(3u << 16),                             // wg_mask=0, data_size=8B
                     (int)((n8 & 0xffffu) << 16),                 // tensor_dim0[15:0]
                     (int)((n8 >> 16) | (1u << 16)),              // tensor_dim0[31:16], tensor_dim1[15:0]=1
                     (int)((n8 & 0xffffu) << 16),                 // tensor_dim1[31:16]=0, tile_dim0=n8
                     1,                                           // tile_dim1=1, tile_dim2=0
                     (int)n8,                                     // tensor_dim0_stride[31:0]
                     0, 0 };
        i32x4 gz = { 0, 0, 0, 0 };
#if __clang_major__ >= 23
        i32x8 gz8 = { 0, 0, 0, 0, 0, 0, 0, 0 };
        __builtin_amdgcn_tensor_load_to_lds(g0, g1, gz, gz, gz8, 0);
#else
        __builtin_amdgcn_tensor_load_to_lds(g0, g1, gz, gz, 0);
#endif
        __builtin_amdgcn_s_wait_tensorcnt(0);
    }
    __syncthreads();
#else
    for (unsigned i = threadIdx.x * 4; i < ndwords; i += blockDim.x * 4)
        *(v4u*)(lds + i) = *(const v4u*)(gsrc + i);
    __syncthreads();
#endif
}

// B fragment (32x16 bf16, KxN), N = pixel: lane n<16 -> col n, K k0..k0+15
// (2 per VGPR, contiguous); lanes 16-31 -> K k0+16..k0+31.
// NHWC bf16 => one contiguous 32-byte read per lane.
__device__ __forceinline__ void loadBact(Frag16& f, const unsigned short* __restrict__ act,
                                         int chs, long pixIdx, int k0, int half, bool valid) {
    v4u z = {0u, 0u, 0u, 0u};
    f.u[0] = z; f.u[1] = z;
    if (valid) {
        const v4u* p = (const v4u*)(act + (size_t)pixIdx * chs + k0 + half * 16);
        f.u[0] = p[0];
        f.u[1] = p[1];
    }
}

// A fragment (weights) from LDS: 8 contiguous dwords per lane (2x ds_load_b128).
__device__ __forceinline__ void loadAlds(Frag16& f, const unsigned int* lds,
                                         int kb, int mb, int MB, int lane) {
    const v4u* p = (const v4u*)(lds + (size_t)(((kb * MB + mb) * 32 + lane) * 8));
    f.u[0] = p[0]; f.u[1] = p[1];
}

// pack 8 f32 -> 8 bf16 -> one 16B store
__device__ __forceinline__ void store_bf8(unsigned short* __restrict__ dst, const v8f& v) {
    v4u pk;
    pk[0] = f2bf_u(v[0]) | (f2bf_u(v[1]) << 16);
    pk[1] = f2bf_u(v[2]) | (f2bf_u(v[3]) << 16);
    pk[2] = f2bf_u(v[4]) | (f2bf_u(v[5]) << 16);
    pk[3] = f2bf_u(v[6]) | (f2bf_u(v[7]) << 16);
    *(v4u*)dst = pk;
}

// ---------------------------------------------------------------------------
// Weight swizzle into the A-fragment layout (16x32 bf16 MxK):
// lane l<16: M=l, VGPR v: K = kb*32 + 2*v      (pairs K,K+1)
// lane l+16: M=l, VGPR v: K = kb*32 + 16 + 2*v
// GEMM k = tap*Cin + ci ; src flat = m*(Cin*ntaps) + ci*ntaps + tap
// ---------------------------------------------------------------------------
__global__ void swizzleW(const float* __restrict__ w, unsigned int* __restrict__ dst,
                         int Cin, int ntaps, int Cout) {
    int Ktot = Cin * ntaps;
    int Kb = Ktot / 32, MB = Cout / 16;
    int total = Kb * MB * 256;
    for (int i = blockIdx.x * blockDim.x + threadIdx.x; i < total;
         i += gridDim.x * blockDim.x) {
        int v    = i & 7;
        int lane = (i >> 3) & 31;
        int blk  = i >> 8;
        int mb = blk % MB, kb = blk / MB;
        int m  = mb * 16 + (lane & 15);
        int K  = kb * 32 + ((lane >> 4) * 16) + 2 * v;
        int t0 = K / Cin,       c0 = K % Cin;
        int t1 = (K + 1) / Cin, c1 = (K + 1) % Cin;
        unsigned int lo = f2bf_u(w[(size_t)m * Ktot + c0 * ntaps + t0]);
        unsigned int hi = f2bf_u(w[(size_t)m * Ktot + c1 * ntaps + t1]);
        dst[i] = lo | (hi << 16);
    }
}

// image+guidance NCHW f32 -> concat NHWC bf16 (128 ch per pixel)
__global__ void packInputs(const float* __restrict__ img, const float* __restrict__ gui,
                           unsigned short* __restrict__ actIn) {
    size_t total = (size_t)BATCH * CHN * HW;
    for (size_t i = blockIdx.x * (size_t)blockDim.x + threadIdx.x; i < total;
         i += (size_t)gridDim.x * blockDim.x) {
        size_t b = i / ((size_t)CHN * HW);
        size_t r = i - b * (size_t)CHN * HW;
        size_t c = r / HW;
        size_t p = r - c * HW;
        size_t o = (b * HW + p) * 128 + c;
        actIn[o]      = (unsigned short)f2bf_u(img[i]);
        actIn[o + 64] = (unsigned short)f2bf_u(gui[i]);
    }
}

// 2-deep pipelined 4-WMMA sweep over mb=0..3 for one (kb, fb) pair.
__device__ __forceinline__ void gemm4_pipelined(v8f* acc, const unsigned int* lds,
                                                int kb, int MB, int lane, const Frag16& fb) {
    Frag16 fa0, fa1;
    loadAlds(fa0, lds, kb, 0, MB, lane);
    loadAlds(fa1, lds, kb, 1, MB, lane);
    acc[0] = WMMA_BF16(fa0, fb, acc[0]);
    loadAlds(fa0, lds, kb, 2, MB, lane);
    acc[1] = WMMA_BF16(fa1, fb, acc[1]);
    loadAlds(fa1, lds, kb, 3, MB, lane);
    acc[2] = WMMA_BF16(fa0, fb, acc[2]);
    acc[3] = WMMA_BF16(fa1, fb, acc[3]);
}

// ---------------------------------------------------------------------------
// Stage 0: 1x1 conv 128->64 + PReLU.  Wave = 16 px, 4 channel-block accs.
// ---------------------------------------------------------------------------
__global__ void __launch_bounds__(128)
conv1x1_fuse(const unsigned short* __restrict__ actIn, const unsigned int* __restrict__ wf,
             const float* __restrict__ bias, const float* __restrict__ alpha,
             unsigned short* __restrict__ out) {
    extern __shared__ unsigned int wlds[];
    stage_weights(wlds, wf, 4 * 4 * 256);
    int lane = threadIdx.x & 31;
    int wave = threadIdx.x >> 5;
    long p0  = ((long)blockIdx.x * 4 + wave) * 16;
    int half = lane >> 4, pl = lane & 15;
    v8f acc[4] = {};
    #pragma unroll
    for (int kc = 0; kc < 4; ++kc) {
        Frag16 fb;
        loadBact(fb, actIn, 128, p0 + pl, kc * 32, half, true);
        gemm4_pipelined(acc, wlds, kc, 4, lane, fb);
    }
    float al = alpha[0];
    unsigned short* outp = out + (size_t)(p0 + pl) * 64 + half * 8;
    #pragma unroll
    for (int mb = 0; mb < 4; ++mb) {
        v8f v;
        #pragma unroll
        for (int r = 0; r < 8; ++r) {
            float t = acc[mb][r] + bias[mb * 16 + half * 8 + r];
            v[r] = t >= 0.f ? t : al * t;
        }
        store_bf8(outp + mb * 16, v);
    }
}

// ---------------------------------------------------------------------------
// 3x3 conv 64->64 (K=576, tap-major).  PRELU: bf16 NHWC.  RESID: f32 NCHW + image.
// ---------------------------------------------------------------------------
template<bool PRELU, bool RESID>
__global__ void __launch_bounds__(128)
conv3x3_wmma(const unsigned short* __restrict__ act, const unsigned int* __restrict__ wf,
             const float* __restrict__ bias, const float* __restrict__ alpha,
             unsigned short* __restrict__ outBf, float* __restrict__ outF32,
             const float* __restrict__ resid) {
    extern __shared__ unsigned int wlds[];
    stage_weights(wlds, wf, 18 * 4 * 256);
    int lane = threadIdx.x & 31;
    int wave = threadIdx.x >> 5;
    int p0 = (blockIdx.x * 4 + wave) * 16;
    int b  = p0 >> 16;
    int sp = p0 & (HW - 1);
    int y0 = sp >> 8;
    int x0 = sp & 255;
    int half = lane >> 4, pl = lane & 15;
    v8f acc[4] = {};
    for (int tap = 0; tap < 9; ++tap) {
        int dy = tap / 3 - 1, dx = tap % 3 - 1;
        int row = y0 + dy;
        if ((unsigned)row >= (unsigned)HH) continue;   // wave-uniform: EXEC stays all-1
        int col = x0 + pl + dx;
        bool valid = (unsigned)col < (unsigned)WW;     // per-lane zero-pad (data only)
        long pix = (long)b * HW + (long)row * WW + col;
        Frag16 fb0, fb1;
        loadBact(fb0, act, 64, pix, 0,  half, valid);
        loadBact(fb1, act, 64, pix, 32, half, valid);
        gemm4_pipelined(acc, wlds, tap * 2 + 0, 4, lane, fb0);
        gemm4_pipelined(acc, wlds, tap * 2 + 1, 4, lane, fb1);
    }
    if (RESID) {
        #pragma unroll
        for (int mb = 0; mb < 4; ++mb) {
            #pragma unroll
            for (int r = 0; r < 8; ++r) {
                int ch = mb * 16 + half * 8 + r;
                size_t idx = ((size_t)b * CHN + ch) * HW + sp + pl;
                outF32[idx] = acc[mb][r] + bias[ch] + resid[idx];
            }
        }
    } else {
        float al = PRELU ? alpha[0] : 0.f;
        unsigned short* outp = outBf + (size_t)(p0 + pl) * 64 + half * 8;
        #pragma unroll
        for (int mb = 0; mb < 4; ++mb) {
            v8f v;
            #pragma unroll
            for (int r = 0; r < 8; ++r) {
                float t = acc[mb][r] + bias[mb * 16 + half * 8 + r];
                v[r] = (!PRELU || t >= 0.f) ? t : al * t;
            }
            store_bf8(outp + mb * 16, v);
        }
    }
}

// ---------------------------------------------------------------------------
// Stage 5: bi_kernel GEMM (64->576) fused with bias + patch multiply + 9-way
// reduction (bi_kernel tensor -- 302 MB -- never materialized).
// M = q (dynamic-kernel row), N = pixel.  The two B fragments (y activations)
// are loaded ONCE per wave and reused across all 36 q-blocks (72 WMMAs).
// Faithful to the reference quirk: q = g*64+cc -> patch channel q/9, tap q%9.
// ---------------------------------------------------------------------------
__global__ void __launch_bounds__(128)
bik_combine(const unsigned short* __restrict__ yact, const unsigned int* __restrict__ wf,
            const float* __restrict__ b5, const float* __restrict__ image,
            unsigned short* __restrict__ jbfOut) {
    extern __shared__ unsigned int wlds[];
    stage_weights(wlds, wf, 2 * 36 * 256);
    int lane = threadIdx.x & 31;
    int wave = threadIdx.x >> 5;
    int p0 = (blockIdx.x * 4 + wave) * 16;
    int b  = p0 >> 16;
    int sp = p0 & (HW - 1);
    int y0 = sp >> 8, x0 = sp & 255;
    int half = lane >> 4, pl = lane & 15;

    Frag16 fb0, fb1;                         // resident activation fragments
    loadBact(fb0, yact, 64, (long)p0 + pl, 0,  half, true);
    loadBact(fb1, yact, 64, (long)p0 + pl, 32, half, true);

    const float* imgb = image + (size_t)b * CHN * HW;
    v8f jacc[4] = {};
    for (int g = 0; g < 9; ++g) {
        #pragma unroll
        for (int cc = 0; cc < 4; ++cc) {
            int mbq = g * 4 + cc;            // q-block over 576/16 = 36
            Frag16 fa0, fa1;                 // both A fragments in flight before wmma
            loadAlds(fa0, wlds, 0, mbq, 36, lane);
            loadAlds(fa1, wlds, 1, mbq, 36, lane);
            v8f acc = {};
            acc = WMMA_BF16(fa0, fb0, acc);
            acc = WMMA_BF16(fa1, fb1, acc);
            #pragma unroll
            for (int r = 0; r < 8; ++r) {
                int q = mbq * 16 + half * 8 + r;
                float bq = b5[q];
                int csrc = q / 9, t = q - csrc * 9;
                int dy = t / 3 - 1, dx = t % 3 - 1;
                int row = y0 + dy, col = x0 + pl + dx;
                float pv = ((unsigned)row < (unsigned)HH && (unsigned)col < (unsigned)WW)
                             ? imgb[(size_t)csrc * HW + row * WW + col] : 0.f;
                jacc[cc][r] += (acc[r] + bq) * pv;
            }
        }
    }
    unsigned short* outp = jbfOut + (size_t)(p0 + pl) * 64 + half * 8;
    #pragma unroll
    for (int cc = 0; cc < 4; ++cc)
        store_bf8(outp + cc * 16, jacc[cc]);
}

// ---------------------------------------------------------------------------
extern "C" void kernel_launch(void* const* d_in, const int* in_sizes, int n_in,
                              void* d_out, int out_size, void* d_ws, size_t ws_size,
                              hipStream_t stream) {
    const float* image    = (const float*)d_in[0];
    const float* guidance = (const float*)d_in[1];
    const float* tensor_w = (const float*)d_in[2];
    const float* tensor_b = (const float*)d_in[3];
    const float* a_jbf    = (const float*)d_in[4];
    const float* kg_w1 = (const float*)d_in[5];  const float* kg_b1 = (const float*)d_in[6];
    const float* kg_w2 = (const float*)d_in[7];  const float* kg_b2 = (const float*)d_in[8];
    const float* kg_w3 = (const float*)d_in[9];  const float* kg_b3 = (const float*)d_in[10];
    const float* kg_w4 = (const float*)d_in[11]; const float* kg_b4 = (const float*)d_in[12];
    const float* kg_w5 = (const float*)d_in[13]; const float* kg_b5 = (const float*)d_in[14];
    const float* a_kg  = (const float*)d_in[15];
    const float* jbf_w1 = (const float*)d_in[16]; const float* jbf_b1 = (const float*)d_in[17];
    const float* jbf_w2 = (const float*)d_in[18]; const float* jbf_b2 = (const float*)d_in[19];
    float* out = (float*)d_out;

    char* ws = (char*)d_ws;
    size_t off = 0;
    auto alloc = [&](size_t bytes) -> void* {
        void* p = ws + off;
        off += (bytes + 255) & ~(size_t)255;
        return p;
    };
    unsigned short* actIn = (unsigned short*)alloc((size_t)BATCH * HW * 128 * 2);
    unsigned short* actA  = (unsigned short*)alloc((size_t)BATCH * HW * 64 * 2);
    unsigned short* actB  = (unsigned short*)alloc((size_t)BATCH * HW * 64 * 2);
    unsigned int* wf0  = (unsigned int*)alloc((size_t)4  * 4  * 256 * 4);   // 128->64, 1x1
    unsigned int* wfk1 = (unsigned int*)alloc((size_t)18 * 4  * 256 * 4);   // 3x3 64->64
    unsigned int* wfk2 = (unsigned int*)alloc((size_t)18 * 4  * 256 * 4);
    unsigned int* wfk3 = (unsigned int*)alloc((size_t)18 * 4  * 256 * 4);
    unsigned int* wfk4 = (unsigned int*)alloc((size_t)18 * 4  * 256 * 4);
    unsigned int* wf5  = (unsigned int*)alloc((size_t)2  * 36 * 256 * 4);   // 64->576, 1x1
    unsigned int* wfj1 = (unsigned int*)alloc((size_t)18 * 4  * 256 * 4);
    unsigned int* wfj2 = (unsigned int*)alloc((size_t)18 * 4  * 256 * 4);

    // --- prep: pack activations, swizzle weights into A-fragment layout ---
    packInputs<<<1024, 256, 0, stream>>>(image, guidance, actIn);
    swizzleW<<<72, 256, 0, stream>>>(tensor_w, wf0, 128, 1, 64);
    swizzleW<<<72, 256, 0, stream>>>(kg_w1, wfk1, 64, 9, 64);
    swizzleW<<<72, 256, 0, stream>>>(kg_w2, wfk2, 64, 9, 64);
    swizzleW<<<72, 256, 0, stream>>>(kg_w3, wfk3, 64, 9, 64);
    swizzleW<<<72, 256, 0, stream>>>(kg_w4, wfk4, 64, 9, 64);
    swizzleW<<<72, 256, 0, stream>>>(kg_w5, wf5, 64, 1, 576);
    swizzleW<<<72, 256, 0, stream>>>(jbf_w1, wfj1, 64, 9, 64);
    swizzleW<<<72, 256, 0, stream>>>(jbf_w2, wfj2, 64, 9, 64);

    const int blocks = BATCH * HW / 64;          // 2048 blocks x 4 waves x 16 px
    const size_t lds1 = 4  * 4 * 256 * 4;        // 16 KB
    const size_t lds3 = 18 * 4 * 256 * 4;        // 72 KB  (4 workgroups / 320 KB WGP)
    const size_t lds5 = 2 * 36 * 256 * 4;        // 72 KB

    // S0: fuse 1x1 (128->64) + PReLU(a_jbf)           -> actA (t)
    conv1x1_fuse<<<blocks, 128, lds1, stream>>>(actIn, wf0, tensor_b, a_jbf, actA);
    // S1..S4: kg 3x3 convs + PReLU(a_kg)
    conv3x3_wmma<true, false><<<blocks, 128, lds3, stream>>>(actA, wfk1, kg_b1, a_kg, actB, nullptr, nullptr);
    conv3x3_wmma<true, false><<<blocks, 128, lds3, stream>>>(actB, wfk2, kg_b2, a_kg, actA, nullptr, nullptr);
    conv3x3_wmma<true, false><<<blocks, 128, lds3, stream>>>(actA, wfk3, kg_b3, a_kg, actB, nullptr, nullptr);
    conv3x3_wmma<true, false><<<blocks, 128, lds3, stream>>>(actB, wfk4, kg_b4, a_kg, actA, nullptr, nullptr);
    // S5: fused bi_kernel GEMM + bias + patch combine -> actB (jbf)
    bik_combine<<<blocks, 128, lds5, stream>>>(actA, wf5, kg_b5, image, actB);
    // S6: jbf 3x3 + PReLU(a_jbf)                      -> actA
    conv3x3_wmma<true, false><<<blocks, 128, lds3, stream>>>(actB, wfj1, jbf_b1, a_jbf, actA, nullptr, nullptr);
    // S7: jbf 3x3 + image residual, f32 NCHW          -> d_out
    conv3x3_wmma<false, true><<<blocks, 128, lds3, stream>>>(actA, wfj2, jbf_b2, nullptr, nullptr, out, image);
}
